// CausalScaledDotAttention_52261162058449
// MI455X (gfx1250) — matrix operand order repaired and checked by reading, not compiled, not run on hardware
//
#include <hip/hip_runtime.h>

// ---------------------------------------------------------------------------
// CausalScaledDotAttention for MI455X (gfx1250, wave32, WMMA bf16 path)
// B=16, S=2048, H=1024
// GEMMs: v_wmma_f32_16x16x32_bf16; B tiles double-buffered in LDS via
// global_load_async_to_lds_b128 (ASYNCcnt) and read back as ds_load_b128.
// ---------------------------------------------------------------------------

#define NB   16
#define SEQ  2048
#define HD   1024
#define NEG_INF_F (-10000000.0f)
#define SCALE_F   (0.03125f)   // 1/sqrt(1024)

#define USE_ASYNC_LDS 1

typedef __attribute__((ext_vector_type(16))) __bf16          v16bf;
typedef __attribute__((ext_vector_type(8)))  float           v8f;
typedef __attribute__((ext_vector_type(16))) unsigned short  ushort16;
typedef __attribute__((ext_vector_type(8)))  unsigned short  ushort8;

__device__ __forceinline__ unsigned short f2bf(float f) {
    unsigned int u = __float_as_uint(f);
    unsigned int r = u + 0x7FFFu + ((u >> 16) & 1u);   // round-to-nearest-even
    return (unsigned short)(r >> 16);
}

// Copy 32 bytes global -> LDS.  Async path uses the CDNA5 async-DMA
// instructions (tracked by ASYNCcnt); both global and LDS addresses get the
// INST_OFFSET, so one address pair serves both halves.
__device__ __forceinline__ void stage32(const unsigned short* __restrict__ g,
                                        unsigned short* __restrict__ l) {
#if USE_ASYNC_LDS
    unsigned long long ga = (unsigned long long)(uintptr_t)g;
    unsigned int       la = (unsigned int)(uintptr_t)l;   // low 32b = LDS offset
    asm volatile("global_load_async_to_lds_b128 %0, %1, off\n\t"
                 "global_load_async_to_lds_b128 %0, %1, off offset:16"
                 :: "v"(la), "v"(ga) : "memory");
#else
    ushort8 t0 = *(const ushort8*)g;
    ushort8 t1 = *(const ushort8*)(g + 8);
    *(ushort8*)l       = t0;
    *(ushort8*)(l + 8) = t1;
#endif
}

// Top-of-iteration sync: my previous async stage is done, then block-wide
// barrier => current buffer fully populated, previous buffer free to restage.
__device__ __forceinline__ void stage_wait() {
#if USE_ASYNC_LDS
    asm volatile("s_wait_asynccnt 0x0" ::: "memory");
#endif
    __syncthreads();
}

// ---------------------------------------------------------------------------
// Kernel 1: convert the three HxH fp32 weight matrices to bf16 in workspace
// ---------------------------------------------------------------------------
__global__ __launch_bounds__(256) void cvt_weights_kernel(
    const float* __restrict__ wq, const float* __restrict__ wk,
    const float* __restrict__ wv,
    unsigned short* __restrict__ wqb, unsigned short* __restrict__ wkb,
    unsigned short* __restrict__ wvb)
{
    size_t i = (size_t)blockIdx.x * blockDim.x + threadIdx.x;   // < H*H
    wqb[i] = f2bf(wq[i]);
    wkb[i] = f2bf(wk[i]);
    wvb[i] = f2bf(wv[i]);
}

// ---------------------------------------------------------------------------
// Kernel 2: projection GEMM  out[s,o] = bf16( sum_h X[s,h]*W[o,h] + bias[o] )
//   M = NB*SEQ, N = K = HD.  Block: 8 waves, tile 128x128; wave tile 16x128.
//   TR_OUT stores the result transposed per batch: out[(b*HD+o)*SEQ + s],
//   used for V so the context GEMM gets a K-contiguous B operand.
// ---------------------------------------------------------------------------
template <bool TR_OUT>
__global__ __launch_bounds__(256) void proj_kernel(
    const float* __restrict__ X, const unsigned short* __restrict__ Wb,
    const float* __restrict__ bias, unsigned short* __restrict__ out)
{
    __shared__ unsigned short btile[2][128 * 32];   // 2 x 8 KB double buffer

    const int lane  = threadIdx.x & 31;
    const int wave  = threadIdx.x >> 5;
    const int m0    = blockIdx.x * 128 + wave * 16;
    const int n0    = blockIdx.y * 128;
    const int mrow  = m0 + (lane & 15);
    const int kbA   = (lane & 16) ? 16 : 0;

    // staging role: thread t copies one 32B run of a W row (n-tile t>>5)
    const size_t srow = (size_t)(n0 + ((threadIdx.x >> 5) * 16) + (lane & 15)) * HD
                        + ((lane & 16) ? 16 : 0);
    unsigned short* sd[2] = { &btile[0][threadIdx.x * 16],
                              &btile[1][threadIdx.x * 16] };

    v8f c[8];
#pragma unroll
    for (int i = 0; i < 8; ++i) c[i] = {};

    stage32(Wb + srow, sd[0]);                       // prologue: stage k0=0
    const int nIter = HD / 32;
    for (int it = 0; it < nIter; ++it) {
        stage_wait();
        if (it + 1 < nIter)                          // stage next K-step into
            stage32(Wb + srow + (it + 1) * 32, sd[(it + 1) & 1]);   // other buf

        const float* ap = X + (size_t)mrow * HD + it * 32 + kbA;
        v8f f0 = *(const v8f*)(ap);
        v8f f1 = *(const v8f*)(ap + 8);
        ushort16 ua;
#pragma unroll
        for (int i = 0; i < 8; ++i) { ua[i] = f2bf(f0[i]); ua[i + 8] = f2bf(f1[i]); }
        v16bf a = __builtin_bit_cast(v16bf, ua);

        v16bf bf[8];
#pragma unroll
        for (int nt = 0; nt < 8; ++nt)
            bf[nt] = *(const v16bf*)&btile[it & 1][(nt * 32 + lane) * 16];
#pragma unroll
        for (int nt = 0; nt < 8; ++nt)
            c[nt] = __builtin_amdgcn_wmma_f32_16x16x32_bf16(
                        false, a, false, bf[nt], (short)0, c[nt], false, false);
    }

    const int rowHi = (lane & 16) ? 8 : 0;
#pragma unroll
    for (int nt = 0; nt < 8; ++nt) {
        const int col = n0 + nt * 16 + (lane & 15);
        const float bv = bias[col];
#pragma unroll
        for (int r = 0; r < 8; ++r) {
            const int row = m0 + r + rowHi;
            unsigned short h = f2bf(c[nt][r] + bv);
            if (TR_OUT) {
                // batch = row>>11, s = row&2047  (SEQ == 2048)
                out[((size_t)((row >> 11) * HD + col)) * SEQ + (row & (SEQ - 1))] = h;
            } else {
                out[(size_t)row * HD + col] = h;
            }
        }
    }
}

// ---------------------------------------------------------------------------
// Kernel 3: scores[b,q,k] = scale*<qhat[b,q,:],khat[b,k,:]> + causal mask
//   (mask adds NEG_INF where q <= k).  Fully-masked tiles (except the last
//   column block, which holds the fully-masked column k=S-1) skip the GEMM
//   and write NEG_INF: downstream exp() underflows to 0.0f identically.
// ---------------------------------------------------------------------------
__global__ __launch_bounds__(256) void scores_kernel(
    const unsigned short* __restrict__ qh, const unsigned short* __restrict__ kh,
    float* __restrict__ attn)
{
    __shared__ unsigned short btile[2][128 * 32];

    const int b     = blockIdx.z;
    const int lane  = threadIdx.x & 31;
    const int wave  = threadIdx.x >> 5;
    const int bm0   = blockIdx.x * 128;               // q tile base (block)
    const int m0    = bm0 + wave * 16;
    const int n0    = blockIdx.y * 128;               // k tile base
    const int rowHi = (lane & 16) ? 8 : 0;
    float* ob = attn + (size_t)b * SEQ * SEQ;

    if (bm0 + 127 <= n0 && n0 + 128 < SEQ) {
        // tile entirely masked and not in the last column block
#pragma unroll
        for (int nt = 0; nt < 8; ++nt) {
            const int kcol = n0 + nt * 16 + (lane & 15);
#pragma unroll
            for (int r = 0; r < 8; ++r)
                ob[(size_t)(m0 + r + rowHi) * SEQ + kcol] = NEG_INF_F;
        }
        return;
    }

    const int qrow = m0 + (lane & 15);
    const int kbA  = (lane & 16) ? 16 : 0;
    const size_t srow = (size_t)(b * SEQ + n0 + ((threadIdx.x >> 5) * 16) + (lane & 15)) * HD
                        + ((lane & 16) ? 16 : 0);
    unsigned short* sd[2] = { &btile[0][threadIdx.x * 16],
                              &btile[1][threadIdx.x * 16] };
    const unsigned short* arow = qh + (size_t)(b * SEQ + qrow) * HD + kbA;

    v8f c[8];
#pragma unroll
    for (int i = 0; i < 8; ++i) c[i] = {};

    stage32(kh + srow, sd[0]);
    const int nIter = HD / 32;
    for (int it = 0; it < nIter; ++it) {
        stage_wait();
        if (it + 1 < nIter)
            stage32(kh + srow + (it + 1) * 32, sd[(it + 1) & 1]);

        const unsigned short* ap = arow + it * 32;
        ushort8 a0 = *(const ushort8*)(ap);
        ushort8 a1 = *(const ushort8*)(ap + 8);
        ushort16 ua;
#pragma unroll
        for (int i = 0; i < 8; ++i) { ua[i] = a0[i]; ua[i + 8] = a1[i]; }
        v16bf a = __builtin_bit_cast(v16bf, ua);

        v16bf bf[8];
#pragma unroll
        for (int nt = 0; nt < 8; ++nt)
            bf[nt] = *(const v16bf*)&btile[it & 1][(nt * 32 + lane) * 16];
#pragma unroll
        for (int nt = 0; nt < 8; ++nt)
            c[nt] = __builtin_amdgcn_wmma_f32_16x16x32_bf16(
                        false, a, false, bf[nt], (short)0, c[nt], false, false);
    }

#pragma unroll
    for (int nt = 0; nt < 8; ++nt) {
        const int kcol = n0 + nt * 16 + (lane & 15);
#pragma unroll
        for (int r = 0; r < 8; ++r) {
            const int q = m0 + r + rowHi;
            float v = c[nt][r] * SCALE_F + ((q <= kcol) ? NEG_INF_F : 0.0f);
            ob[(size_t)q * SEQ + kcol] = v;
        }
    }
}

// ---------------------------------------------------------------------------
// Kernel 4: softmax over the QUERY axis (axis=1 of (B,q,k)) per (b,k) column.
// ---------------------------------------------------------------------------
__global__ __launch_bounds__(256) void softmax_q_kernel(float* __restrict__ attn)
{
    const int g = blockIdx.x * 256 + threadIdx.x;     // column id over NB*SEQ
    const int b = g / SEQ;
    const int k = g % SEQ;
    float* p = attn + (size_t)b * SEQ * SEQ + k;

    float m = -3.4e38f, s = 0.0f;
    for (int q = 0; q < SEQ; ++q) {
        float x  = p[(size_t)q * SEQ];
        float nm = fmaxf(m, x);
        s = s * __expf(m - nm) + __expf(x - nm);
        m = nm;
    }
    const float inv = 1.0f / s;
    for (int q = 0; q < SEQ; ++q) {
        size_t idx = (size_t)q * SEQ;
        p[idx] = __expf(p[idx] - m) * inv;
    }
}

// ---------------------------------------------------------------------------
// Kernel 5: context[b,q,h] = sum_k attn[b,q,k] * v[b,k,h]
//   A = attn fp32 (K-contiguous, cvt to bf16), B = vhT[b][h][s] (K-contiguous
//   thanks to the transposed V projection).  attn[q,k] == 0.0f exactly for
//   k >= q except the fully-masked column k=S-1, so the K loop covers only
//   [0, diag_block_end) plus the final 32-wide block.
// ---------------------------------------------------------------------------
__global__ __launch_bounds__(256) void context_kernel(
    const float* __restrict__ attn, const unsigned short* __restrict__ vhT,
    float* __restrict__ ctx)
{
    __shared__ unsigned short btile[2][128 * 32];

    const int b     = blockIdx.z;
    const int lane  = threadIdx.x & 31;
    const int wave  = threadIdx.x >> 5;
    const int bm0   = blockIdx.x * 128;               // q tile base (block)
    const int m0    = bm0 + wave * 16;
    const int n0    = blockIdx.y * 128;               // h tile base
    const int qrow  = m0 + (lane & 15);
    const int kbA   = (lane & 16) ? 16 : 0;

    const size_t srow = (size_t)(b * HD + n0 + ((threadIdx.x >> 5) * 16) + (lane & 15)) * SEQ
                        + ((lane & 16) ? 16 : 0);
    unsigned short* sd[2] = { &btile[0][threadIdx.x * 16],
                              &btile[1][threadIdx.x * 16] };
    const float* arow = attn + (size_t)b * SEQ * SEQ + (size_t)qrow * SEQ + kbA;

    v8f c[8];
#pragma unroll
    for (int i = 0; i < 8; ++i) c[i] = {};

    // K-step schedule: blocks up to the diagonal, plus the final block
    // (columns S-32..S-1) which holds the fully-masked column S-1.
    const int nDiag = (bm0 + 128) >> 5;
    const int nIter = nDiag + (((bm0 + 128) < SEQ) ? 1 : 0);
    auto kof = [&](int i) { return (i < nDiag) ? (i << 5) : (SEQ - 32); };

    stage32(vhT + srow + kof(0), sd[0]);
    for (int it = 0; it < nIter; ++it) {
        stage_wait();
        if (it + 1 < nIter)
            stage32(vhT + srow + kof(it + 1), sd[(it + 1) & 1]);

        const float* ap = arow + kof(it);
        v8f f0 = *(const v8f*)(ap);
        v8f f1 = *(const v8f*)(ap + 8);
        ushort16 ua;
#pragma unroll
        for (int i = 0; i < 8; ++i) { ua[i] = f2bf(f0[i]); ua[i + 8] = f2bf(f1[i]); }
        v16bf a = __builtin_bit_cast(v16bf, ua);

        v16bf bf[8];
#pragma unroll
        for (int nt = 0; nt < 8; ++nt)
            bf[nt] = *(const v16bf*)&btile[it & 1][(nt * 32 + lane) * 16];
#pragma unroll
        for (int nt = 0; nt < 8; ++nt)
            c[nt] = __builtin_amdgcn_wmma_f32_16x16x32_bf16(
                        false, a, false, bf[nt], (short)0, c[nt], false, false);
    }

    const int rowHi = (lane & 16) ? 8 : 0;
    float* ob = ctx + (size_t)b * SEQ * HD;
#pragma unroll
    for (int nt = 0; nt < 8; ++nt) {
        const int h = n0 + nt * 16 + (lane & 15);
#pragma unroll
        for (int r = 0; r < 8; ++r) {
            const int q = m0 + r + rowHi;
            ob[(size_t)q * HD + h] = c[nt][r];
        }
    }
}

// ---------------------------------------------------------------------------
extern "C" void kernel_launch(void* const* d_in, const int* in_sizes, int n_in,
                              void* d_out, int out_size, void* d_ws, size_t ws_size,
                              hipStream_t stream)
{
    const float* queries = (const float*)d_in[0];
    const float* keys    = (const float*)d_in[1];
    const float* values  = (const float*)d_in[2];
    const float* Wq      = (const float*)d_in[3];
    const float* bq      = (const float*)d_in[4];
    const float* Wk      = (const float*)d_in[5];
    const float* bk      = (const float*)d_in[6];
    const float* Wv      = (const float*)d_in[7];
    const float* bv      = (const float*)d_in[8];

    float* ctx  = (float*)d_out;                                  // (B,S,H)
    float* attn = (float*)d_out + (size_t)NB * SEQ * HD;          // (B,S,S)

    char* ws = (char*)d_ws;
    unsigned short* Wqb = (unsigned short*)(ws);
    unsigned short* Wkb = (unsigned short*)(ws + (size_t)HD * HD * 2);
    unsigned short* Wvb = (unsigned short*)(ws + (size_t)HD * HD * 4);
    unsigned short* qh  = (unsigned short*)(ws + (size_t)HD * HD * 6);
    unsigned short* kh  = qh + (size_t)NB * SEQ * HD;
    unsigned short* vhT = kh + (size_t)NB * SEQ * HD;

    // 1. weights -> bf16
    cvt_weights_kernel<<<(HD * HD) / 256, 256, 0, stream>>>(
        Wq, Wk, Wv, Wqb, Wkb, Wvb);

    // 2. projections (M = NB*SEQ = 32768, N = K = HD); V stored transposed
    dim3 pgrid((NB * SEQ) / 128, HD / 128);
    proj_kernel<false><<<pgrid, 256, 0, stream>>>(queries, Wqb, bq, qh);
    proj_kernel<false><<<pgrid, 256, 0, stream>>>(keys,    Wkb, bk, kh);
    proj_kernel<true ><<<pgrid, 256, 0, stream>>>(values,  Wvb, bv, vhT);

    // 3. masked scaled scores -> attn buffer (upper tiles short-circuited)
    scores_kernel<<<dim3(SEQ / 128, SEQ / 128, NB), 256, 0, stream>>>(qh, kh, attn);

    // 4. softmax over the query axis, in place
    softmax_q_kernel<<<(NB * SEQ) / 256, 256, 0, stream>>>(attn);

    // 5. context = attn @ v   (K loop causally truncated)
    context_kernel<<<dim3(SEQ / 128, HD / 128, NB), 256, 0, stream>>>(attn, vhT, ctx);
}
